// GraphEmbedding2_33586644254938
// MI455X (gfx1250) — matrix operand order repaired and checked by reading, not compiled, not run on hardware
//
#include <hip/hip_runtime.h>
#include <hip/hip_bf16.h>

// ---------------------------------------------------------------------------
// Problem constants (from reference): N=16, T=48, H=1024, E=16384, P=4,
// D=128, G=8, S=128, tp=T/P=12, B=N*tp=192, BH=B*H=196608.
// ---------------------------------------------------------------------------
#define CN   16
#define CT   48
#define CH   1024
#define CE   16384
#define CP   4
#define CD   128
#define CG   8
#define CS   128
#define CTP  (CT / CP)         // 12
#define CB   (CN * CTP)        // 192
#define CBH  (CB * CH)         // 196608
#define CEH  (CE + CH)         // 17408 total (edges + self loops)

typedef __attribute__((ext_vector_type(16))) _Float16 v16h;
typedef __attribute__((ext_vector_type(8)))  _Float16 v8h;
typedef __attribute__((ext_vector_type(8)))  float    v8f;

// ---------------------------------------------------------------------------
// Graph preprocessing: degree -> scan -> CSR fill
// ---------------------------------------------------------------------------
__global__ void k_init_deg(int* __restrict__ deg) {
    int i = blockIdx.x * blockDim.x + threadIdx.x;
    if (i < CH) deg[i] = 1;                   // self loop
}

__global__ void k_count_deg(const int* __restrict__ edges, int* __restrict__ deg) {
    int e = blockIdx.x * blockDim.x + threadIdx.x;
    if (e < CE) atomicAdd(&deg[edges[2 * e + 1]], 1);
}

// One block of 1024 threads: inclusive scan of deg -> rowStart/cursor, dinv.
__global__ __launch_bounds__(1024) void k_scan(const int* __restrict__ deg,
                                               int* __restrict__ rowStart,
                                               int* __restrict__ cursor,
                                               float* __restrict__ dinv) {
    __shared__ int sh[CH];
    const int tid = threadIdx.x;
    const int v = deg[tid];
    sh[tid] = v;
    __syncthreads();
    #pragma unroll
    for (int off = 1; off < CH; off <<= 1) {
        int t = (tid >= off) ? sh[tid - off] : 0;
        __syncthreads();
        sh[tid] += t;
        __syncthreads();
    }
    rowStart[tid + 1] = sh[tid];
    if (tid == 0) rowStart[0] = 0;
    cursor[tid] = sh[tid] - v;                // exclusive
    dinv[tid] = rsqrtf((float)v);
}

__global__ void k_fill_csr(const int* __restrict__ edges,
                           const float* __restrict__ dinv,
                           int* __restrict__ cursor,
                           int* __restrict__ csr_src,
                           float* __restrict__ csr_w) {
    int i = blockIdx.x * blockDim.x + threadIdx.x;
    if (i >= CEH) return;
    int src, dst;
    if (i < CE) { src = edges[2 * i]; dst = edges[2 * i + 1]; }
    else        { src = i - CE; dst = src; }   // self loop
    int pos = atomicAdd(&cursor[dst], 1);
    csr_src[pos] = src;
    csr_w[pos]   = dinv[src] * dinv[dst];
}

// ---------------------------------------------------------------------------
// Weight convert: row-major f32 W[k][j] -> column-major f16 Wc[j*128 + k]
// ---------------------------------------------------------------------------
__global__ void k_cvt_w(const float* __restrict__ W, _Float16* __restrict__ Wc) {
    int i = blockIdx.x * blockDim.x + threadIdx.x;   // 16384 elements
    int k = i >> 7, j = i & 127;
    Wc[j * CD + k] = (_Float16)W[k * CD + j];
}

// ---------------------------------------------------------------------------
// Conv embedding: emb[(n*tp+t)*H + h, d] = b[d] + sum_p x[n, 4t+p, h]*w[d,p]
// Output directly in f16 (feeds the WMMA GEMM).
// ---------------------------------------------------------------------------
__global__ __launch_bounds__(256) void k_conv_emb(const float* __restrict__ x,
                                                  const float* __restrict__ cw,
                                                  const float* __restrict__ cb,
                                                  _Float16* __restrict__ emb) {
    const int idx = blockIdx.x * 256 + threadIdx.x;  // over CBH*CD
    const int d = idx & 127;
    const int r = idx >> 7;                           // row in [0, CBH)
    const int h = r & (CH - 1);
    const int b = r >> 10;                            // n*tp + t
    const int n = b / CTP;
    const int t = b % CTP;
    const float* xp = x + (size_t)n * (CT * CH) + (size_t)(t * CP) * CH + h;
    float acc = cb[d];
    #pragma unroll
    for (int p = 0; p < CP; ++p) acc += xp[p * CH] * cw[d * CP + p];
    emb[(size_t)r * CD + d] = (_Float16)acc;
}

// ---------------------------------------------------------------------------
// WMMA GEMM: Y(M,128) = X(M,128) @ W(128,128).  X,Y,W in f16, f32 accumulate.
// Wc is column-major. One workgroup = 8 waves = 128 rows.
// Per wave: 16-row strip, 8 N-tiles x 4 K-tiles = 32 v_wmma_f32_16x16x32_f16.
// ---------------------------------------------------------------------------
__global__ __launch_bounds__(256) void k_gemm_f16_128(const _Float16* __restrict__ X,
                                                      const _Float16* __restrict__ Wc,
                                                      _Float16* __restrict__ Y) {
    __shared__ __attribute__((aligned(32))) _Float16 lw[CD * CD];  // 32 KB
    {   // cooperative load of column-major W into LDS (b128 copies)
        const v8h* s = (const v8h*)Wc;
        v8h* dst = (v8h*)lw;
        #pragma unroll
        for (int i = threadIdx.x; i < (CD * CD) / 8; i += 256) dst[i] = s[i];
    }
    __syncthreads();

    const int wave = threadIdx.x >> 5;
    const int lane = threadIdx.x & 31;
    const int lgrp = lane >> 4;      // 0 or 1 (K half selector)
    const int lidx = lane & 15;      // row (A) / column (B/C/D)
    const size_t rowBase = (size_t)blockIdx.x * 128 + wave * 16;
    const _Float16* arow = X + (rowBase + lidx) * CD;

    // A fragments for all 4 K-tiles (contiguous K order, matching B):
    //   lane<16 : halfs0..15 = K kt*32+0..15
    //   lane>=16: halfs0..15 = K kt*32+16..31
    v16h a[4];
    #pragma unroll
    for (int kt = 0; kt < 4; ++kt) {
        const int koff = kt * 32 + lgrp * 16;
        a[kt] = *(const v16h*)(arow + koff);
    }

    _Float16* yout = Y + rowBase * CD;
    #pragma unroll
    for (int nt = 0; nt < 8; ++nt) {
        v8f acc = {};
        #pragma unroll
        for (int kt = 0; kt < 4; ++kt) {
            // B 32x16 layout: lane<16 holds K kt*32+0..15 of col lidx,
            // lane>=16 holds K kt*32+16..31 -> contiguous in column-major LDS.
            const v16h bf = *(const v16h*)&lw[(nt * 16 + lidx) * CD + kt * 32 + lgrp * 16];
            acc = __builtin_amdgcn_wmma_f32_16x16x32_f16(
                false, a[kt], false, bf, (short)0, acc, false, false);
        }
        // C/D layout: VGPR r -> row r (lanes 0-15) / row r+8 (lanes 16-31)
        #pragma unroll
        for (int r = 0; r < 8; ++r) {
            const int row = r + lgrp * 8;
            yout[(size_t)row * CD + nt * 16 + lidx] = (_Float16)acc[r];
        }
    }
}

// ---------------------------------------------------------------------------
// GCN aggregation (gather form, deterministic):
//   out[b,dst,d] = bias[d] + sum_{e in CSR[dst]} w_e * hw[b, src_e, d]
// optional ReLU. One 128-thread block per (b, dst); coalesced 256B row reads.
// ---------------------------------------------------------------------------
__global__ __launch_bounds__(128) void k_aggregate(const _Float16* __restrict__ HW,
                                                   const int* __restrict__ csr_src,
                                                   const float* __restrict__ csr_w,
                                                   const int* __restrict__ rowStart,
                                                   const float* __restrict__ bias,
                                                   _Float16* __restrict__ OUT,
                                                   int do_relu) {
    const int dst = blockIdx.x & (CH - 1);
    const int b   = blockIdx.x >> 10;
    const int d   = threadIdx.x;
    const _Float16* base = HW + (size_t)b * CH * CD;
    const int s = rowStart[dst], e = rowStart[dst + 1];
    float acc = 0.f;
    for (int i = s; i < e; ++i) {
        const int src = csr_src[i];
        const float w = csr_w[i];
        acc += w * (float)base[(size_t)src * CD + d];
    }
    acc += bias[d];
    if (do_relu) acc = fmaxf(acc, 0.f);
    OUT[(size_t)blockIdx.x * CD + d] = (_Float16)acc;
}

// ---------------------------------------------------------------------------
// Grouped mean pool: out[(b*G+g)*D + d] = mean_s tok[b, split[g,s], d]
// ---------------------------------------------------------------------------
__global__ __launch_bounds__(128) void k_group_mean(const _Float16* __restrict__ tok,
                                                    const int* __restrict__ split,
                                                    float* __restrict__ out) {
    const int g = blockIdx.x & (CG - 1);
    const int b = blockIdx.x >> 3;
    const int d = threadIdx.x;
    const _Float16* base = tok + (size_t)b * CH * CD;
    float acc = 0.f;
    #pragma unroll 4
    for (int s = 0; s < CS; ++s) {
        const int node = split[g * CS + s];
        acc += (float)base[(size_t)node * CD + d];
    }
    out[(size_t)blockIdx.x * CD + d] = acc * (1.0f / CS);
}

// ---------------------------------------------------------------------------
// Launcher
// ---------------------------------------------------------------------------
extern "C" void kernel_launch(void* const* d_in, const int* in_sizes, int n_in,
                              void* d_out, int out_size, void* d_ws, size_t ws_size,
                              hipStream_t stream) {
    const float* x      = (const float*)d_in[0];
    const int*   edges  = (const int*)  d_in[1];
    const int*   split  = (const int*)  d_in[2];
    const float* conv_w = (const float*)d_in[3];
    const float* conv_b = (const float*)d_in[4];
    const float* W1     = (const float*)d_in[5];
    const float* b1     = (const float*)d_in[6];
    const float* W2     = (const float*)d_in[7];
    const float* b2     = (const float*)d_in[8];
    float* out = (float*)d_out;

    // Workspace layout (two 50.3MB f16 ping-pong buffers + small region)
    char* ws = (char*)d_ws;
    const size_t FEAT = (size_t)CBH * CD * sizeof(_Float16);   // 50,331,648
    _Float16* bufA = (_Float16*)(ws);
    _Float16* bufB = (_Float16*)(ws + FEAT);
    char* sm = ws + 2 * FEAT;
    int*      deg      = (int*)(sm);                 sm += 4096;
    float*    dinv     = (float*)(sm);               sm += 4096;
    int*      rowStart = (int*)(sm);                 sm += 4352;   // 1025 ints
    int*      cursor   = (int*)(sm);                 sm += 4096;
    int*      csr_src  = (int*)(sm);                 sm += CEH * 4;
    float*    csr_w    = (float*)(sm);               sm += CEH * 4;
    _Float16* Wc1      = (_Float16*)(sm);            sm += CD * CD * 2;
    _Float16* Wc2      = (_Float16*)(sm);            sm += CD * CD * 2;
    (void)ws_size; (void)in_sizes; (void)n_in; (void)out_size;

    // Graph preprocessing (cheap, rebuilt every call -> deterministic)
    k_init_deg <<<(CH + 255) / 256, 256, 0, stream>>>(deg);
    k_count_deg<<<(CE + 255) / 256, 256, 0, stream>>>(edges, deg);
    k_scan     <<<1, 1024, 0, stream>>>(deg, rowStart, cursor, dinv);
    k_fill_csr <<<(CEH + 255) / 256, 256, 0, stream>>>(edges, dinv, cursor, csr_src, csr_w);

    // Weight conversion to column-major f16
    k_cvt_w<<<(CD * CD) / 256, 256, 0, stream>>>(W1, Wc1);
    k_cvt_w<<<(CD * CD) / 256, 256, 0, stream>>>(W2, Wc2);

    // Conv embedding -> bufA (f16)
    k_conv_emb<<<((size_t)CBH * CD) / 256, 256, 0, stream>>>(x, conv_w, conv_b, bufA);

    // Layer 1: hw1 = emb @ W1 (WMMA) -> bufB; hid = relu(agg + b1) -> bufA
    k_gemm_f16_128<<<CBH / 128, 256, 0, stream>>>(bufA, Wc1, bufB);
    k_aggregate   <<<CBH, 128, 0, stream>>>(bufB, csr_src, csr_w, rowStart, b1, bufA, 1);

    // Layer 2: hw2 = hid @ W2 (WMMA) -> bufB; tok = agg + b2 -> bufA
    k_gemm_f16_128<<<CBH / 128, 256, 0, stream>>>(bufA, Wc2, bufB);
    k_aggregate   <<<CBH, 128, 0, stream>>>(bufB, csr_src, csr_w, rowStart, b2, bufA, 0);

    // Grouped mean pool -> out (f32)
    k_group_mean<<<CB * CG, 128, 0, stream>>>(bufA, split, out);
}
